// ECForGraphTCN_12532714570020
// MI455X (gfx1250) — compile-verified, matrix operand order checked by reading, not run on hardware
//
#include <hip/hip_runtime.h>
#include <hip/hip_bf16.h>
#include <hip/hip_fp16.h>

// ---------------------------------------------------------------------------
// MI455X (gfx1250): all MLPs on v_wmma_f32_16x16x32_f16 (wave32).
// Weights pre-swizzled into WMMA B-register layout -> 2x global_load_b128
// per fragment. A-fragments read as 2x ds_load_b128 from LDS tiles.
// Gathers use shift/mask addressing only (no integer division).
// ---------------------------------------------------------------------------

typedef __attribute__((ext_vector_type(16))) _Float16 v16h;
typedef __attribute__((ext_vector_type(8)))  _Float16 v8h;
typedef __attribute__((ext_vector_type(8)))  float    v8f;
using half_t = _Float16;

#define TPB   256
#define WAVES 8
#define ROWS  64

// ---- A fragment (16x32 f16, MxK) from row-major LDS tile ------------------
__device__ __forceinline__ v16h load_a_frag(const half_t* __restrict__ A, int lda,
                                            int mBase, int kBase, int lane) {
  const half_t* p = A + (mBase + (lane & 15)) * lda + kBase + ((lane >> 4) << 3);
  v8h lo = *(const v8h*)(p);
  v8h hi = *(const v8h*)(p + 16);
  return __builtin_shufflevector(lo, hi, 0, 1, 2, 3, 4, 5, 6, 7,
                                 8, 9, 10, 11, 12, 13, 14, 15);
}

// ---- B fragment from pre-swizzled weights ---------------------------------
// Swizzled layout: tile (kt, nt) of (32K x 16N) stored as 512 contiguous
// halves: dst[(kt*(N/16)+nt)*512 + lane*16 + i]  (i in WMMA register order).
template<int K, int NOUT>
__device__ __forceinline__ v8f wmma_tile(const half_t* __restrict__ A, int lda,
                                         const half_t* __restrict__ Bsw,
                                         int mBase, int nt, int lane) {
  v8f acc = {};
#pragma unroll
  for (int k0 = 0; k0 < K; k0 += 32) {
    v16h a = load_a_frag(A, lda, mBase, k0, lane);
    const v16h b = *(const v16h*)(Bsw +
        (((size_t)((k0 >> 5) * (NOUT >> 4) + nt)) << 9) + ((size_t)lane << 4));
    acc = __builtin_amdgcn_wmma_f32_16x16x32_f16(false, a, false, b,
                                                 (short)0, acc, false, false);
  }
  return acc;
}

// Full hidden stage: act[64][NOUT] = relu(A[64][K] @ W[K][NOUT] + bias)
template<int K, int NOUT>
__device__ __forceinline__ void mlp_stage_relu(const half_t* __restrict__ A, int lda,
                                               const half_t* __restrict__ Wsw,
                                               const float* __restrict__ bias,
                                               half_t* __restrict__ act,
                                               int wave, int lane) {
  constexpr int NT = (ROWS / 16) * (NOUT / 16);
  for (int t = wave; t < NT; t += WAVES) {
    const int mt = t & 3, nt = t >> 2;
    v8f acc = wmma_tile<K, NOUT>(A, lda, Wsw, mt * 16, nt, lane);
    const int col = nt * 16 + (lane & 15);
    const float bv = bias ? bias[col] : 0.f;
    const int rb = mt * 16 + ((lane >> 4) << 3);  // C/D: M = r (+8 upper lanes)
#pragma unroll
    for (int r = 0; r < 8; ++r)
      act[(rb + r) * NOUT + col] = (half_t)fmaxf(acc[r] + bv, 0.f);
  }
}

// ---- weight fp32 -> fp16 conversion + WMMA B-layout swizzle ---------------
__global__ void convert_w_kernel(const float* __restrict__ src, half_t* __restrict__ dst,
                                 int Kin, int Kpad, int Ncols) {
  int i = blockIdx.x * blockDim.x + threadIdx.x;
  if (i >= Kpad * Ncols) return;
  const int e    = i & 15;         // element within lane (register order)
  const int lane = (i >> 4) & 31;
  const int tile = i >> 9;
  const int ntN  = Ncols >> 4;
  const int kt = tile / ntN, nt = tile - kt * ntN;
  const int k = kt * 32 + ((e & 8) ? 16 : 0) + ((lane >> 4) << 3) + (e & 7);
  const int n = nt * 16 + (lane & 15);
  dst[i] = (k < Kin) ? (half_t)src[k * Ncols + n] : (half_t)0.f;
}

__global__ void zero_kernel(float* __restrict__ p, long long n) {
  long long i = (long long)blockIdx.x * blockDim.x + threadIdx.x;
  if (i < n) p[i] = 0.f;
}

// ---- encoder: in[rows,KIN] -> relu(x@W0) -> relu(@W1) ---------------------
template<int KIN, int NOUT, bool WF32>
__global__ __launch_bounds__(TPB)
void encoder_kernel(const float* __restrict__ in,
                    const half_t* __restrict__ W0, const half_t* __restrict__ W1,
                    half_t* __restrict__ outh, float* __restrict__ outf, int nrows) {
  __shared__ __align__(16) half_t A[ROWS * 32];
  __shared__ __align__(16) half_t act1[ROWS * 128];
  const int tid = threadIdx.x, lane = tid & 31, wave = tid >> 5;
  const long long rBase = (long long)blockIdx.x * ROWS;

  for (int i = tid; i < ROWS * 32; i += TPB) {
    int row = i >> 5, k = i & 31;
    long long rr = rBase + row;
    float v = (k < KIN && rr < nrows) ? in[rr * KIN + k] : 0.f;
    A[i] = (half_t)v;
  }
  __syncthreads();
  mlp_stage_relu<32, 128>(A, 32, W0, nullptr, act1, wave, lane);
  __syncthreads();

  constexpr int NT = (ROWS / 16) * (NOUT / 16);
  for (int t = wave; t < NT; t += WAVES) {
    const int mt = t & 3, nt = t >> 2;
    v8f acc = wmma_tile<128, NOUT>(act1, 128, W1, mt * 16, nt, lane);
    const int col = nt * 16 + (lane & 15);
    const int rb = mt * 16 + ((lane >> 4) << 3);
#pragma unroll
    for (int r = 0; r < 8; ++r) {
      long long rr = rBase + rb + r;
      if (rr < nrows) {
        float v = fmaxf(acc[r], 0.f);              // outer relu wrap
        outh[rr * NOUT + col] = (half_t)v;
        if (WF32) outf[rr * NOUT + col] = v;
      }
    }
  }
}

// ---- rel MLP (per edge), fused gather + segment-sum scatter ---------------
__global__ __launch_bounds__(TPB)
void rel_kernel(const half_t* __restrict__ hf, const half_t* __restrict__ ef_in,
                const int* __restrict__ srcIdx, const int* __restrict__ dstIdx,
                const half_t* __restrict__ W0, const float* __restrict__ b0,
                const half_t* __restrict__ W1, const float* __restrict__ b1,
                const half_t* __restrict__ W2, const float* __restrict__ b2,
                half_t* __restrict__ ef_out, float* __restrict__ agg, int E) {
  __shared__ __align__(16) half_t A[ROWS * 160];     // [h_dst | h_src | e]
  __shared__ __align__(16) half_t act1[ROWS * 128];
  __shared__ __align__(16) half_t act2[ROWS * 128];
  __shared__ int sSrc[ROWS];
  __shared__ int sDst[ROWS];
  const int tid = threadIdx.x, lane = tid & 31, wave = tid >> 5;
  const long long eBase = (long long)blockIdx.x * ROWS;

  if (tid < ROWS) {
    long long e = eBase + tid;
    sSrc[tid] = (e < E) ? srcIdx[e] : 0;
    sDst[tid] = (e < E) ? dstIdx[e] : -1;
  }
  __syncthreads();

  // Segment-wise gather, shift/mask addressing only.
  unsigned* Au = (unsigned*)A;                   // row stride 80 dwords
  const unsigned* hfu = (const unsigned*)hf;     // 32 dwords per node row
  const unsigned* efu = (const unsigned*)ef_in;  // 16 dwords per edge row
  for (int i = tid; i < ROWS * 32; i += TPB) {   // h[dst]
    int row = i >> 5, s = i & 31;
    int d = sDst[row];
    Au[row * 80 + s] = (d >= 0) ? hfu[(size_t)d * 32 + s] : 0u;
  }
  for (int i = tid; i < ROWS * 32; i += TPB) {   // h[src]
    int row = i >> 5, s = i & 31;
    Au[row * 80 + 32 + s] = (sDst[row] >= 0) ? hfu[(size_t)sSrc[row] * 32 + s] : 0u;
  }
  for (int i = tid; i < ROWS * 16; i += TPB) {   // e
    int row = i >> 4, s = i & 15;
    Au[row * 80 + 64 + s] = (sDst[row] >= 0) ? efu[(eBase + row) * 16 + s] : 0u;
  }
  __syncthreads();

  mlp_stage_relu<160, 128>(A, 160, W0, b0, act1, wave, lane);
  __syncthreads();
  mlp_stage_relu<128, 128>(act1, 128, W1, b1, act2, wave, lane);
  __syncthreads();

  // final 128->32, no relu; fused e-store + segment_sum atomics
  for (int t = wave; t < 8; t += WAVES) {
    const int mt = t & 3, nt = t >> 2;
    v8f acc = wmma_tile<128, 32>(act2, 128, W2, mt * 16, nt, lane);
    const int col = nt * 16 + (lane & 15);
    const float bv = b2[col];
    const int rb = mt * 16 + ((lane >> 4) << 3);
#pragma unroll
    for (int r = 0; r < 8; ++r) {
      const int row = rb + r;
      long long e = eBase + row;
      if (e < E) {
        float v = acc[r] + bv;
        ef_out[e * 32 + col] = (half_t)v;
        atomicAdd(&agg[(size_t)sDst[row] * 32 + col], v);   // segment_sum
      }
    }
  }
}

// ---- obj MLP (per node) + convex residual update --------------------------
__global__ __launch_bounds__(TPB)
void obj_kernel(half_t* __restrict__ hf, const float* __restrict__ agg,
                const half_t* __restrict__ W0, const float* __restrict__ b0,
                const half_t* __restrict__ W1, const float* __restrict__ b1,
                const half_t* __restrict__ W2, const float* __restrict__ b2,
                float* __restrict__ h, int N) {
  __shared__ __align__(16) half_t A[ROWS * 96];     // [h | agg]
  __shared__ __align__(16) half_t act1[ROWS * 128];
  __shared__ __align__(16) half_t act2[ROWS * 128];
  const int tid = threadIdx.x, lane = tid & 31, wave = tid >> 5;
  const long long nBase = (long long)blockIdx.x * ROWS;

  unsigned* Au = (unsigned*)A;                  // row stride 48 dwords
  const unsigned* hfu = (const unsigned*)hf;
  for (int i = tid; i < ROWS * 32; i += TPB) {  // h half (vectorized dwords)
    int row = i >> 5, s = i & 31;
    long long node = nBase + row;
    Au[row * 48 + s] = (node < N) ? hfu[node * 32 + s] : 0u;
  }
  for (int i = tid; i < ROWS * 32; i += TPB) {  // agg half (f32 -> f16)
    int row = i >> 5, c = i & 31;
    long long node = nBase + row;
    float v = (node < N) ? agg[node * 32 + c] : 0.f;
    A[row * 96 + 64 + c] = (half_t)v;
  }
  __syncthreads();

  mlp_stage_relu<96, 128>(A, 96, W0, b0, act1, wave, lane);
  __syncthreads();
  mlp_stage_relu<128, 128>(act1, 128, W1, b1, act2, wave, lane);
  __syncthreads();

  for (int t = wave; t < 16; t += WAVES) {   // 128 -> 64
    const int mt = t & 3, nt = t >> 2;
    v8f acc = wmma_tile<128, 64>(act2, 128, W2, mt * 16, nt, lane);
    const int col = nt * 16 + (lane & 15);
    const float bv = b2[col];
    const int rb = mt * 16 + ((lane >> 4) << 3);
#pragma unroll
    for (int r = 0; r < 8; ++r) {
      long long node = nBase + rb + r;
      if (node < N) {
        float delta = fmaxf(acc[r] + bv, 0.f);
        long long o = node * 64 + col;
        float hn = 0.5f * h[o] + 0.5f * delta;   // ALPHA = 0.5
        h[o] = hn;
        hf[o] = (half_t)hn;
      }
    }
  }
}

// ---- edge head over concat of 4 e-buffers ---------------------------------
__global__ __launch_bounds__(TPB)
void head_kernel(const half_t* __restrict__ e0, const half_t* __restrict__ e1,
                 const half_t* __restrict__ e2, const half_t* __restrict__ e3,
                 const half_t* __restrict__ W0, const float* __restrict__ b0,
                 const half_t* __restrict__ W1, const float* __restrict__ b1,
                 const float* __restrict__ w2, const float* __restrict__ b2,
                 float* __restrict__ out, int E) {
  __shared__ __align__(16) half_t A[ROWS * 128];
  __shared__ __align__(16) half_t act1[ROWS * 128];
  __shared__ __align__(16) half_t act2[ROWS * 128];
  __shared__ float red[TPB];
  const int tid = threadIdx.x, lane = tid & 31, wave = tid >> 5;
  const long long eBase = (long long)blockIdx.x * ROWS;

  unsigned* Au = (unsigned*)A;
  for (int i = tid; i < ROWS * 64; i += TPB) {   // 64 dwords per row
    int row = i >> 6, s = i & 63;
    long long e = eBase + row;
    unsigned v = 0u;
    if (e < E) {
      const unsigned* p = (s < 16) ? (const unsigned*)e0
                        : (s < 32) ? (const unsigned*)e1
                        : (s < 48) ? (const unsigned*)e2
                                   : (const unsigned*)e3;
      v = p[e * 16 + (s & 15)];
    }
    Au[i] = v;
  }
  __syncthreads();

  mlp_stage_relu<128, 128>(A, 128, W0, b0, act1, wave, lane);
  __syncthreads();
  mlp_stage_relu<128, 128>(act1, 128, W1, b1, act2, wave, lane);
  __syncthreads();

  // final 128 -> 1 dot: 4 partial sums of 32 per row, LDS reduce
  {
    const int row = tid >> 2, part = tid & 3;
    const half_t* ap = act2 + row * 128 + part * 32;
    const float*  wp = w2 + part * 32;
    float partial = 0.f;
#pragma unroll
    for (int k = 0; k < 32; ++k) partial += (float)ap[k] * wp[k];
    red[tid] = partial;
  }
  __syncthreads();
  if (tid < ROWS) {
    long long e = eBase + tid;
    if (e < E) {
      const int b = tid << 2;
      float s = red[b] + red[b + 1] + red[b + 2] + red[b + 3] + b2[0];
      out[e] = 1.f / (1.f + __expf(-s));
    }
  }
}

// ---------------------------------------------------------------------------
extern "C" void kernel_launch(void* const* d_in, const int* in_sizes, int n_in,
                              void* d_out, int out_size, void* d_ws, size_t ws_size,
                              hipStream_t stream) {
  (void)in_sizes; (void)n_in; (void)out_size; (void)ws_size;
  const int N = 50000, E = 1600000;

  // ---- input mapping (pytree flatten order; bias=None dropped) ----
  const float* x   = (const float*)d_in[0];
  const float* ea  = (const float*)d_in[1];
  const float* nW0 = (const float*)d_in[2];   // [14,128]
  const float* nW1 = (const float*)d_in[3];   // [128,64]
  const float* eW0 = (const float*)d_in[4];   // [4,128]
  const float* eW1 = (const float*)d_in[5];   // [128,32]
  struct LP { const float *rW0,*rb0,*rW1,*rb1,*rW2,*rb2,*oW0,*ob0,*oW1,*ob1,*oW2,*ob2; } L[3];
  for (int i = 0; i < 3; ++i) {
    int b = 6 + 12 * i;
    L[i] = { (const float*)d_in[b+0], (const float*)d_in[b+1], (const float*)d_in[b+2],
             (const float*)d_in[b+3], (const float*)d_in[b+4], (const float*)d_in[b+5],
             (const float*)d_in[b+6], (const float*)d_in[b+7], (const float*)d_in[b+8],
             (const float*)d_in[b+9], (const float*)d_in[b+10],(const float*)d_in[b+11] };
  }
  const float* wW0 = (const float*)d_in[42];
  const float* wb0 = (const float*)d_in[43];
  const float* wW1 = (const float*)d_in[44];
  const float* wb1 = (const float*)d_in[45];
  const float* wW2 = (const float*)d_in[46];  // [128,1] kept fp32
  const float* wb2 = (const float*)d_in[47];
  const int* eidx  = (const int*)d_in[48];
  const int* srcI  = eidx;
  const int* dstI  = eidx + E;

  // ---- workspace bump allocator ----
  char* ws = (char*)d_ws;
  size_t off = 0;
  auto alloc = [&](size_t bytes) -> void* {
    off = (off + 255) & ~(size_t)255;
    void* p = ws + off; off += bytes; return p;
  };
  half_t* hf   = (half_t*)alloc((size_t)N * 64 * 2);
  float*  h    = (float*) alloc((size_t)N * 64 * 4);
  float*  agg  = (float*) alloc((size_t)N * 32 * 4);
  half_t* ef[4];
  for (int i = 0; i < 4; ++i) ef[i] = (half_t*)alloc((size_t)E * 32 * 2);
  half_t* nW0h = (half_t*)alloc(32 * 128 * 2);
  half_t* nW1h = (half_t*)alloc(128 * 64 * 2);
  half_t* eW0h = (half_t*)alloc(32 * 128 * 2);
  half_t* eW1h = (half_t*)alloc(128 * 32 * 2);
  half_t *rW0h[3], *rW1h[3], *rW2h[3], *oW0h[3], *oW1h[3], *oW2h[3];
  for (int i = 0; i < 3; ++i) {
    rW0h[i] = (half_t*)alloc(160 * 128 * 2);
    rW1h[i] = (half_t*)alloc(128 * 128 * 2);
    rW2h[i] = (half_t*)alloc(128 * 32 * 2);
    oW0h[i] = (half_t*)alloc(96 * 128 * 2);
    oW1h[i] = (half_t*)alloc(128 * 128 * 2);
    oW2h[i] = (half_t*)alloc(128 * 64 * 2);
  }
  half_t* wW0h = (half_t*)alloc(128 * 128 * 2);
  half_t* wW1h = (half_t*)alloc(128 * 128 * 2);

  auto conv = [&](const float* s, half_t* d, int kin, int kpad, int ncol) {
    int tot = kpad * ncol;
    convert_w_kernel<<<(tot + 255) / 256, 256, 0, stream>>>(s, d, kin, kpad, ncol);
  };
  conv(nW0, nW0h, 14, 32, 128);  conv(nW1, nW1h, 128, 128, 64);
  conv(eW0, eW0h, 4, 32, 128);   conv(eW1, eW1h, 128, 128, 32);
  for (int i = 0; i < 3; ++i) {
    conv(L[i].rW0, rW0h[i], 160, 160, 128);
    conv(L[i].rW1, rW1h[i], 128, 128, 128);
    conv(L[i].rW2, rW2h[i], 128, 128, 32);
    conv(L[i].oW0, oW0h[i], 96, 96, 128);
    conv(L[i].oW1, oW1h[i], 128, 128, 128);
    conv(L[i].oW2, oW2h[i], 128, 128, 64);
  }
  conv(wW0, wW0h, 128, 128, 128);
  conv(wW1, wW1h, 128, 128, 128);

  const int nodeBlocks = (N + ROWS - 1) / ROWS;   // 782
  const int edgeBlocks = (E + ROWS - 1) / ROWS;   // 25000

  encoder_kernel<14, 64, true ><<<nodeBlocks, TPB, 0, stream>>>(x,  nW0h, nW1h, hf,    h,       N);
  encoder_kernel< 4, 32, false><<<edgeBlocks, TPB, 0, stream>>>(ea, eW0h, eW1h, ef[0], nullptr, E);

  for (int l = 0; l < 3; ++l) {
    long long aggN = (long long)N * 32;
    zero_kernel<<<(int)((aggN + 255) / 256), 256, 0, stream>>>(agg, aggN);
    rel_kernel<<<edgeBlocks, TPB, 0, stream>>>(hf, ef[l], srcI, dstI,
        rW0h[l], L[l].rb0, rW1h[l], L[l].rb1, rW2h[l], L[l].rb2,
        ef[l + 1], agg, E);
    obj_kernel<<<nodeBlocks, TPB, 0, stream>>>(hf, agg,
        oW0h[l], L[l].ob0, oW1h[l], L[l].ob1, oW2h[l], L[l].ob2, h, N);
  }

  head_kernel<<<edgeBlocks, TPB, 0, stream>>>(ef[0], ef[1], ef[2], ef[3],
      wW0h, wb0, wW1h, wb1, wW2, wb2, (float*)d_out, E);
}
